// SingleHeadSelfAttention_71794673320159
// MI455X (gfx1250) — matrix operand order repaired and checked by reading, not compiled, not run on hardware
//
#include <hip/hip_runtime.h>
#include <hip/hip_bf16.h>

typedef __attribute__((ext_vector_type(16))) __bf16   v16bf;
typedef __attribute__((ext_vector_type(8)))  __bf16   v8bf;
typedef __attribute__((ext_vector_type(8)))  float    v8f;
typedef __attribute__((ext_vector_type(8)))  _Float16 v8h;

#define D_MODEL 1024
#define SEQ     4096
#define SCALE   0.03125f   // 1/sqrt(1024)

static __device__ __forceinline__ v8f wmma_bf16(v16bf a, v16bf b, v8f c) {
  return __builtin_amdgcn_wmma_f32_16x16x32_bf16(false, a, false, b, (short)0, c,
                                                 false, false);
}

// A-fragment (16x32 bf16): lane = 16*half + m holds row M=m.
// elements 0..7  <- K in [kb + 16*half, +8); 8..15 <- K in [kb + 16*half + 8, +8)
static __device__ __forceinline__ v16bf load_a_bf16(const __bf16* row, int kb, int half) {
  v8bf lo = *(const v8bf*)(row + kb + 16 * half);
  v8bf hi = *(const v8bf*)(row + kb + 16 * half + 8);
  return __builtin_shufflevector(lo, hi, 0, 1, 2, 3, 4, 5, 6, 7,
                                         8, 9, 10, 11, 12, 13, 14, 15);
}

// B-fragment (32x16 bf16): lane holds column N=m; element h -> K = kb + 16*half + h.
static __device__ __forceinline__ v16bf load_b_bf16(const __bf16* row, int kb, int half) {
  return *(const v16bf*)(row + kb + 16 * half);
}

// ---------------------------------------------------------------------------
// Kernel 0: bulk fp32 -> bf16 conversion (8 elements per thread).
// ---------------------------------------------------------------------------
__global__ void __launch_bounds__(256)
cvt_kernel(const float* __restrict__ src, __bf16* __restrict__ dst, int n8) {
  const int i = blockIdx.x * 256 + threadIdx.x;
  if (i >= n8) return;
  const size_t base = (size_t)i * 8;
  float4 a = *(const float4*)(src + base);
  float4 b = *(const float4*)(src + base + 4);
  v8bf o;
  o[0] = (__bf16)a.x; o[1] = (__bf16)a.y; o[2] = (__bf16)a.z; o[3] = (__bf16)a.w;
  o[4] = (__bf16)b.x; o[5] = (__bf16)b.y; o[6] = (__bf16)b.z; o[7] = (__bf16)b.w;
  *(v8bf*)(dst + base) = o;
}

// ---------------------------------------------------------------------------
// Kernel 1: Q/K/V projections (bf16 x bf16 -> bf16), triple-buffered pipeline.
// One wave per 16x64 output tile.  V written transposed: Vt[b][e][k].
// ---------------------------------------------------------------------------
__global__ void __launch_bounds__(256)
qkv_kernel(const __bf16* __restrict__ Xb,  const __bf16* __restrict__ Wqb,
           const __bf16* __restrict__ Wkb, const __bf16* __restrict__ Wvb,
           __bf16* __restrict__ Qb, __bf16* __restrict__ Kb, __bf16* __restrict__ Vt) {
  const int lane = threadIdx.x & 31;
  const int wid  = blockIdx.x * 8 + (threadIdx.x >> 5);
  const int half = lane >> 4, m = lane & 15;
  const int mat  = wid >> 13;          // 0=Q, 1=K, 2=V
  const int t    = wid & 8191;
  const int n0   = (t >> 4) * 16;
  const int e0   = (t & 15) * 64;
  const __bf16* W  = (mat == 0) ? Wqb : (mat == 1) ? Wkb : Wvb;
  const __bf16* xr = Xb + (size_t)(n0 + m) * D_MODEL;
  const __bf16* wr = W + (size_t)(e0 + m) * D_MODEL;   // j offsets become immediates

  v8f acc[4] = {};
  v16bf ab[3], bb[3][4];
  ab[0] = load_a_bf16(xr, 0, half);
#pragma unroll
  for (int j = 0; j < 4; ++j) bb[0][j] = load_b_bf16(wr + (size_t)j * 16 * D_MODEL, 0, half);
  ab[1] = load_a_bf16(xr, 32, half);
#pragma unroll
  for (int j = 0; j < 4; ++j) bb[1][j] = load_b_bf16(wr + (size_t)j * 16 * D_MODEL, 32, half);

#pragma unroll
  for (int s = 0; s < 32; ++s) {           // 32 k-steps of 32
    const int cur = s % 3;
    if (s + 2 < 32) {
      const int nxt = (s + 2) % 3;
      ab[nxt] = load_a_bf16(xr, (s + 2) * 32, half);
#pragma unroll
      for (int j = 0; j < 4; ++j)
        bb[nxt][j] = load_b_bf16(wr + (size_t)j * 16 * D_MODEL, (s + 2) * 32, half);
    }
#pragma unroll
    for (int j = 0; j < 4; ++j) acc[j] = wmma_bf16(ab[cur], bb[cur][j], acc[j]);
  }

  if (mat < 2) {
    __bf16* O = (mat == 0) ? Qb : Kb;
#pragma unroll
    for (int j = 0; j < 4; ++j)
#pragma unroll
      for (int r = 0; r < 8; ++r)
        O[(size_t)(n0 + r + 8 * half) * D_MODEL + e0 + j * 16 + m] = (__bf16)acc[j][r];
  } else {
    const int b   = n0 >> 12;
    const int nb0 = n0 & 4095;
#pragma unroll
    for (int j = 0; j < 4; ++j) {
      v8bf v;
#pragma unroll
      for (int r = 0; r < 8; ++r) v[r] = (__bf16)acc[j][r];
      *(v8bf*)(Vt + (size_t)b * D_MODEL * SEQ +
               (size_t)(e0 + j * 16 + m) * SEQ + nb0 + 8 * half) = v;
    }
  }
}

// ---------------------------------------------------------------------------
// Kernel 2: scores = (Q K^T) * scale, softmax rows -> P (bf16).
// Workgroup: 16 query rows x all 4096 keys; 16 waves, 16 key tiles each.
// Triple-buffered WMMA sweep; scores stashed in registers as packed f16.
// ---------------------------------------------------------------------------
__global__ void __launch_bounds__(512)
scores_softmax_kernel(const __bf16* __restrict__ Qb, const __bf16* __restrict__ Kb,
                      __bf16* __restrict__ P) {
  const int b    = blockIdx.y;
  const int q0   = blockIdx.x * 16;
  const int w    = threadIdx.x >> 5;
  const int lane = threadIdx.x & 31;
  const int half = lane >> 4, m = lane & 15;
  const __bf16* qrow = Qb + (size_t)(b * SEQ + q0 + m) * D_MODEL;

  __shared__ float smax[16][16];
  __shared__ float ssum[16][16];

  float rm[8], rs[8];
  v8h   sreg[16];
#pragma unroll
  for (int r = 0; r < 8; ++r) { rm[r] = -1e30f; rs[r] = 0.f; }

  for (int i = 0; i < 16; ++i) {
    const int k0 = (w + i * 16) * 16;
    const __bf16* krow = Kb + (size_t)(b * SEQ + k0 + m) * D_MODEL;

    v8f acc = {};
    v16bf ab[3], bb[3];
    ab[0] = load_a_bf16(qrow, 0, half);
    bb[0] = load_b_bf16(krow, 0, half);
    ab[1] = load_a_bf16(qrow, 32, half);
    bb[1] = load_b_bf16(krow, 32, half);
#pragma unroll
    for (int s = 0; s < 32; ++s) {
      const int cur = s % 3;
      if (s + 2 < 32) {
        const int nxt = (s + 2) % 3;
        ab[nxt] = load_a_bf16(qrow, (s + 2) * 32, half);
        bb[nxt] = load_b_bf16(krow, (s + 2) * 32, half);
      }
      acc = wmma_bf16(ab[cur], bb[cur], acc);
    }

    float s[8], tmax[8], tsum[8];
#pragma unroll
    for (int r = 0; r < 8; ++r) {
      s[r] = acc[r] * SCALE;
      sreg[i][r] = (_Float16)s[r];
      tmax[r] = s[r];
    }
#pragma unroll
    for (int mask = 8; mask >= 1; mask >>= 1)
#pragma unroll
      for (int r = 0; r < 8; ++r)
        tmax[r] = fmaxf(tmax[r], __shfl_xor(tmax[r], mask, 32));
#pragma unroll
    for (int r = 0; r < 8; ++r) tsum[r] = __expf(s[r] - tmax[r]);
#pragma unroll
    for (int mask = 8; mask >= 1; mask >>= 1)
#pragma unroll
      for (int r = 0; r < 8; ++r) tsum[r] += __shfl_xor(tsum[r], mask, 32);
#pragma unroll
    for (int r = 0; r < 8; ++r) {
      float nm = fmaxf(rm[r], tmax[r]);
      rs[r] = rs[r] * __expf(rm[r] - nm) + tsum[r] * __expf(tmax[r] - nm);
      rm[r] = nm;
    }
  }

  if (m == 0) {
#pragma unroll
    for (int r = 0; r < 8; ++r) {
      smax[w][r + 8 * half] = rm[r];
      ssum[w][r + 8 * half] = rs[r];
    }
  }
  __syncthreads();

  float M[8], invL[8];
#pragma unroll
  for (int r = 0; r < 8; ++r) {
    const int row = r + 8 * half;
    float mm = -1e30f;
    for (int ww = 0; ww < 16; ++ww) mm = fmaxf(mm, smax[ww][row]);
    float L = 0.f;
    for (int ww = 0; ww < 16; ++ww) L += ssum[ww][row] * __expf(smax[ww][row] - mm);
    M[r] = mm;
    invL[r] = 1.0f / L;
  }

  for (int i = 0; i < 16; ++i) {
    const int k0 = (w + i * 16) * 16;
#pragma unroll
    for (int r = 0; r < 8; ++r) {
      float p = __expf((float)sreg[i][r] - M[r]) * invL[r];
      P[(size_t)b * SEQ * SEQ + (size_t)(q0 + r + 8 * half) * SEQ + k0 + m] = (__bf16)p;
    }
  }
}

// ---------------------------------------------------------------------------
// Kernel 3: O = P @ V (bf16 -> fp32), wave per 16x128 tile, ping-pong pipelined.
// ---------------------------------------------------------------------------
__global__ void __launch_bounds__(256)
pv_kernel(const __bf16* __restrict__ P, const __bf16* __restrict__ Vt,
          float* __restrict__ O) {
  const int lane = threadIdx.x & 31;
  const int wid  = blockIdx.x * 8 + (threadIdx.x >> 5);
  const int half = lane >> 4, m = lane & 15;
  const int b    = wid >> 11;
  const int t    = wid & 2047;
  const int q0   = (t >> 3) * 16;
  const int e0   = (t & 7) * 128;
  const __bf16* prow  = P  + (size_t)b * SEQ * SEQ + (size_t)(q0 + m) * SEQ;
  const __bf16* vrow  = Vt + (size_t)b * D_MODEL * SEQ + (size_t)(e0 + m) * SEQ;

  v8f acc[8] = {};
  v16bf a0, a1, b0[8], b1[8];
  a0 = load_a_bf16(prow, 0, half);
#pragma unroll
  for (int j = 0; j < 8; ++j) b0[j] = load_b_bf16(vrow + (size_t)j * 16 * SEQ, 0, half);

  for (int kb = 0; kb < SEQ - 64; kb += 64) {
    a1 = load_a_bf16(prow, kb + 32, half);
#pragma unroll
    for (int j = 0; j < 8; ++j)
      b1[j] = load_b_bf16(vrow + (size_t)j * 16 * SEQ, kb + 32, half);
#pragma unroll
    for (int j = 0; j < 8; ++j) acc[j] = wmma_bf16(a0, b0[j], acc[j]);

    a0 = load_a_bf16(prow, kb + 64, half);
#pragma unroll
    for (int j = 0; j < 8; ++j)
      b0[j] = load_b_bf16(vrow + (size_t)j * 16 * SEQ, kb + 64, half);
#pragma unroll
    for (int j = 0; j < 8; ++j) acc[j] = wmma_bf16(a1, b1[j], acc[j]);
  }
  a1 = load_a_bf16(prow, SEQ - 32, half);
#pragma unroll
  for (int j = 0; j < 8; ++j)
    b1[j] = load_b_bf16(vrow + (size_t)j * 16 * SEQ, SEQ - 32, half);
#pragma unroll
  for (int j = 0; j < 8; ++j) acc[j] = wmma_bf16(a0, b0[j], acc[j]);
#pragma unroll
  for (int j = 0; j < 8; ++j) acc[j] = wmma_bf16(a1, b1[j], acc[j]);

#pragma unroll
  for (int j = 0; j < 8; ++j)
#pragma unroll
    for (int r = 0; r < 8; ++r)
      O[(size_t)(b * SEQ + q0 + r + 8 * half) * D_MODEL + e0 + j * 16 + m] = acc[j][r];
}

// ---------------------------------------------------------------------------
extern "C" void kernel_launch(void* const* d_in, const int* in_sizes, int n_in,
                              void* d_out, int out_size, void* d_ws, size_t ws_size,
                              hipStream_t stream) {
  (void)in_sizes; (void)n_in; (void)out_size; (void)ws_size;
  const float* X  = (const float*)d_in[0];
  const float* Wq = (const float*)d_in[1];
  const float* Wk = (const float*)d_in[2];
  const float* Wv = (const float*)d_in[3];
  float* out = (float*)d_out;

  const size_t MB = 1024 * 1024;
  char* ws = (char*)d_ws;
  __bf16* Xb  = (__bf16*)(ws);              // 16 MB
  __bf16* Wqb = (__bf16*)(ws + 16 * MB);    //  2 MB
  __bf16* Wkb = (__bf16*)(ws + 18 * MB);    //  2 MB
  __bf16* Wvb = (__bf16*)(ws + 20 * MB);    //  2 MB
  __bf16* Qb  = (__bf16*)(ws + 24 * MB);    // 16 MB
  __bf16* Kb  = (__bf16*)(ws + 40 * MB);    // 16 MB
  __bf16* Vt  = (__bf16*)(ws + 56 * MB);    // 16 MB
  __bf16* P   = (__bf16*)(ws + 72 * MB);    // 64 MB

  hipLaunchKernelGGL(cvt_kernel, dim3(4096), dim3(256), 0, stream, X,  Xb,  1048576);
  hipLaunchKernelGGL(cvt_kernel, dim3(512),  dim3(256), 0, stream, Wq, Wqb, 131072);
  hipLaunchKernelGGL(cvt_kernel, dim3(512),  dim3(256), 0, stream, Wk, Wkb, 131072);
  hipLaunchKernelGGL(cvt_kernel, dim3(512),  dim3(256), 0, stream, Wv, Wvb, 131072);

  hipLaunchKernelGGL(qkv_kernel, dim3(3072), dim3(256), 0, stream,
                     Xb, Wqb, Wkb, Wvb, Qb, Kb, Vt);
  hipLaunchKernelGGL(scores_softmax_kernel, dim3(256, 2), dim3(512), 0, stream,
                     Qb, Kb, P);
  hipLaunchKernelGGL(pv_kernel, dim3(512), dim3(256), 0, stream, P, Vt, out);
}
